// BiDAF_GPU_22582938042782
// MI455X (gfx1250) — matrix operand order repaired and checked by reading, not compiled, not run on hardware
//
#include <hip/hip_runtime.h>
#include <hip/hip_bf16.h>

// ---------------------------------------------------------------------------
// BiDAF forward for MI455X (gfx1250, wave32).
// GEMMs: f16 WMMA (v_wmma_f32_16x16x32_f16), f32 accumulate, LDS-tiled,
//        b128 async global->LDS tile fill (ASYNCcnt). Wave tile 16x64
//        (1 A-frag reused across 4 B-frags -> 4 WMMAs per K-step).
// Recurrent scans: 1 workgroup / direction, h in LDS, VALU dots (serial part).
// All f16 GEMM operands are K-padded to multiples of 32 so tile loads are
// branch-free; M/N edges handled by index clamp + guarded store.
// ---------------------------------------------------------------------------

typedef __attribute__((ext_vector_type(16))) _Float16 v16h;
typedef __attribute__((ext_vector_type(8)))  float    v8f;
typedef __attribute__((ext_vector_type(4)))  int      b128i;

#define TM 64
#define TN 128
#define TK 32

#if __has_builtin(__builtin_amdgcn_global_load_async_to_lds_b128)
#define HAS_ASYNC_LDS 1
#else
#define HAS_ASYNC_LDS 0
#endif

static __device__ __forceinline__ int imin(int a, int b) { return a < b ? a : b; }

// 16-byte global -> LDS copy (async if available)
static __device__ __forceinline__ void copy16_g2l(_Float16* lds, const _Float16* g)
{
#if HAS_ASYNC_LDS
    __builtin_amdgcn_global_load_async_to_lds_b128(
        (__attribute__((address_space(1))) b128i*)(g),
        (__attribute__((address_space(3))) b128i*)(lds), 0, 0);
#else
    *(float4*)lds = *(const float4*)g;
#endif
}
static __device__ __forceinline__ void wait_async_tiles()
{
#if HAS_ASYNC_LDS
#if __has_builtin(__builtin_amdgcn_s_wait_asynccnt)
    __builtin_amdgcn_s_wait_asynccnt(0);
#else
    asm volatile("s_wait_asynccnt 0" ::: "memory");
#endif
#endif
}

// C[M,N] f32 = A[M,K]f16 @ op(B)f16 (+bias).
//   BT=1: B is (N-rows, ldb) row-major -> A@B^T        (vector tile fill)
//   BT=0: B is (K-rows, ldb) row-major -> A@B          (scalar transpose fill)
// K must be a multiple of 32 (operands zero-padded in K).
// Block: 256 thr = 8 waves as 4(M) x 2(N); wave tile 16 x 64.
template <int BT>
__global__ __launch_bounds__(256)
void gemm_wmma(const _Float16* __restrict__ A, int lda,
               const _Float16* __restrict__ B, int ldb,
               const float* __restrict__ bias, float* __restrict__ Cout,
               int M, int N, int K)
{
    __shared__ _Float16 As[TM][TK + 8];
    __shared__ _Float16 Bs[TN][TK + 8];
    const int tid  = threadIdx.x;
    const int lane = tid & 31;
    const int wave = tid >> 5;          // 8 waves
    const int wm   = wave >> 1;         // 0..3  (16-row subtile)
    const int wn   = wave & 1;          // 0..1  (64-col half)
    const int m0 = blockIdx.y * TM;
    const int n0 = blockIdx.x * TN;

    // tile-fill coordinates: one 8xf16 vector per load
    const int fr  = tid >> 2;           // 0..63
    const int fc8 = (tid & 3) * 8;      // 0,8,16,24

    // hoisted global pointers (advance by TK per iteration)
    const _Float16* aptr  = A + (size_t)imin(m0 + fr, M - 1) * lda + fc8;
    const _Float16* bptr0 = B + (size_t)imin(n0 + fr, N - 1) * ldb + fc8;        // BT=1
    const _Float16* bptr1 = B + (size_t)imin(n0 + 64 + fr, N - 1) * ldb + fc8;   // BT=1

    v8f acc0 = {}, acc1 = {}, acc2 = {}, acc3 = {};
    for (int k0 = 0; k0 < K; k0 += TK) {
        if (BT) {
            copy16_g2l(&Bs[fr][fc8],      bptr0);
            copy16_g2l(&Bs[64 + fr][fc8], bptr1);
            bptr0 += TK; bptr1 += TK;
        } else {
            int gn0 = imin(n0 + fr, N - 1);
            int gn1 = imin(n0 + 64 + fr, N - 1);
#pragma unroll
            for (int kk = 0; kk < 8; ++kk) {
                Bs[fr][fc8 + kk]      = B[(size_t)(k0 + fc8 + kk) * ldb + gn0];
                Bs[64 + fr][fc8 + kk] = B[(size_t)(k0 + fc8 + kk) * ldb + gn1];
            }
        }
        copy16_g2l(&As[fr][fc8], aptr);
        aptr += TK;

        wait_async_tiles();
        __syncthreads();

        // ---- fragment gather (contiguous k = 16*hi + j, same order for A and B) ----
        const int hi  = (lane >= 16) ? 1 : 0;
        const int row = lane & 15;
        v16h af, bf0, bf1, bf2, bf3;
#pragma unroll
        for (int j = 0; j < 16; ++j) {
            int ka = j + hi * 16;                    // A: K index for element j
            af[j] = As[wm * 16 + row][ka];
            int kb = j + hi * 16;                    // B: K index for element j
            bf0[j] = Bs[wn * 64 +  0 + row][kb];
            bf1[j] = Bs[wn * 64 + 16 + row][kb];
            bf2[j] = Bs[wn * 64 + 32 + row][kb];
            bf3[j] = Bs[wn * 64 + 48 + row][kb];
        }
        acc0 = __builtin_amdgcn_wmma_f32_16x16x32_f16(false, af, false, bf0, (short)0, acc0, false, false);
        acc1 = __builtin_amdgcn_wmma_f32_16x16x32_f16(false, af, false, bf1, (short)0, acc1, false, false);
        acc2 = __builtin_amdgcn_wmma_f32_16x16x32_f16(false, af, false, bf2, (short)0, acc2, false, false);
        acc3 = __builtin_amdgcn_wmma_f32_16x16x32_f16(false, af, false, bf3, (short)0, acc3, false, false);
        __syncthreads();
    }

    // ---- store C/D (VGPR r: M = r + (lane>=16 ? 8 : 0), N = lane&15) ----
    const int col = lane & 15;
    const int mh  = (lane >= 16) ? 8 : 0;
    v8f* accs[4] = { &acc0, &acc1, &acc2, &acc3 };
#pragma unroll
    for (int t = 0; t < 4; ++t) {
        int gnb = n0 + wn * 64 + t * 16 + col;
#pragma unroll
        for (int r = 0; r < 8; ++r) {
            int gm = m0 + wm * 16 + r + mh;
            if (gm < M && gnb < N) {
                float v = (*accs[t])[r];
                if (bias) v += bias[gnb];
                Cout[(size_t)gm * N + gnb] = v;
            }
        }
    }
}

// ---------------------------------------------------------------------------
// Recurrent LSTM scan: blockIdx.x = direction (0 fwd, 1 bwd). h in LDS.
// out[t*712 + dir*356 + j] = h_j(t)   (bwd writes at original index).
// ---------------------------------------------------------------------------
__global__ __launch_bounds__(384)
void lstm_scan(const float* __restrict__ xW0, const float* __restrict__ xW1,
               const _Float16* __restrict__ Whh,   // (2,1424,356) f16
               const float* __restrict__ h0,       // (2,356)
               float* __restrict__ out,            // (T,712)
               int T)
{
    const int H = 356;
    const int dir = blockIdx.x;
    const float* xW = dir ? xW1 : xW0;
    const _Float16* W = Whh + (size_t)dir * 1424 * H;
    __shared__ float hs[356];
    const int j = threadIdx.x;
    float c = 0.f, h = 0.f;
    if (j < H) { h = h0[dir * H + j]; c = h; hs[j] = h; }
    __syncthreads();
    for (int t = 0; t < T; ++t) {
        int tt = dir ? (T - 1 - t) : t;
        float nh = h, nc = c;
        if (j < H) {
            const float* xr = xW + (size_t)tt * 1424;
            float zi = xr[j], zf = xr[H + j], zg = xr[2 * H + j], zo = xr[3 * H + j];
            const _Float16* wi = W + (size_t)j * H;
            const _Float16* wf = W + (size_t)(H + j) * H;
            const _Float16* wg = W + (size_t)(2 * H + j) * H;
            const _Float16* wo = W + (size_t)(3 * H + j) * H;
#pragma unroll 4
            for (int k = 0; k < H; ++k) {
                float hk = hs[k];
                zi += (float)wi[k] * hk;
                zf += (float)wf[k] * hk;
                zg += (float)wg[k] * hk;
                zo += (float)wo[k] * hk;
            }
            float si = 1.f / (1.f + __expf(-zi));
            float sf = 1.f / (1.f + __expf(-zf));
            float so = 1.f / (1.f + __expf(-zo));
            nc = sf * c + si * tanhf(zg);
            nh = so * tanhf(nc);
        }
        __syncthreads();
        if (j < H) {
            c = nc; h = nh;
            hs[j] = h;
            out[(size_t)tt * 712 + dir * H + j] = h;
        }
        __syncthreads();
    }
}

// ---------------------------------------------------------------------------
// Small helper kernels
// ---------------------------------------------------------------------------
__global__ void f32_to_f16(const float* __restrict__ in, _Float16* __restrict__ out, size_t n)
{
    size_t i = (size_t)blockIdx.x * blockDim.x + threadIdx.x;
    if (i < n) out[i] = (_Float16)in[i];
}

// padded f32 -> f16: out[r*Kp + k] = (k < K) ? in[r*K + k] : 0
__global__ void f32_to_f16_pad(const float* __restrict__ in, _Float16* __restrict__ out,
                               int rows, int K, int Kp)
{
    size_t i = (size_t)blockIdx.x * blockDim.x + threadIdx.x;
    if (i >= (size_t)rows * Kp) return;
    int r = (int)(i / Kp), k = (int)(i % Kp);
    out[i] = (k < K) ? (_Float16)in[(size_t)r * K + k] : (_Float16)0.f;
}

// char CNN: y[n,oc] = max_p ( b[oc] + sum_{k,c} emb[chars[n,p+k-2],c]*w[oc,c,k] )
__global__ void char_cnn(const int* __restrict__ chars, const float* __restrict__ emb,
                         const float* __restrict__ w, const float* __restrict__ b,
                         float* __restrict__ out, int N)
{
    int id = blockIdx.x * blockDim.x + threadIdx.x;
    if (id >= N * 100) return;
    int n = id / 100, oc = id % 100;
    float bb = b[oc];
    float best = -3.4e38f;
    for (int p = 0; p < 16; ++p) {
        float s = bb;
        for (int k = 0; k < 5; ++k) {
            int pos = p + k - 2;
            if (pos < 0 || pos >= 16) continue;
            int ch = chars[n * 16 + pos];
            const float* e  = emb + (size_t)ch * 100;
            const float* ww = w + (size_t)oc * 500 + k;  // w[oc][c][k], stride 5 over c
            for (int c = 0; c < 100; ++c) s += e[c] * ww[c * 5];
        }
        best = fmaxf(best, s);
    }
    out[id] = best;
}

// x[n,0:100]=cnn, x[n,100:356]=elmo; x16 is K-padded to Hp (zeros)
__global__ void build_x(const float* __restrict__ cnn, const float* __restrict__ elmo,
                        float* __restrict__ x, _Float16* __restrict__ x16, int T, int Hp)
{
    int i = blockIdx.x * blockDim.x + threadIdx.x;
    if (i >= T * Hp) return;
    int n = i / Hp, d = i % Hp;
    if (d < 356) {
        float v = (d < 100) ? cnn[n * 100 + d] : elmo[(size_t)n * 256 + (d - 100)];
        x[n * 356 + d] = v;
        x16[i] = (_Float16)v;
    } else {
        x16[i] = (_Float16)0.f;
    }
}

// highway combine: x = sigmoid(gz)*relu(pz) + (1-sigmoid(gz))*x ; x16 padded
__global__ void hw_combine(const float* __restrict__ gz, const float* __restrict__ pz,
                           float* __restrict__ x, _Float16* __restrict__ x16, int n, int Hp)
{
    int i = blockIdx.x * blockDim.x + threadIdx.x;
    if (i >= n) return;
    float g = 1.f / (1.f + __expf(-gz[i]));
    float p = fmaxf(pz[i], 0.f);
    float v = g * p + (1.f - g) * x[i];
    x[i] = v;
    x16[(size_t)(i / 356) * Hp + (i % 356)] = (_Float16)v;
}

// out[n*Dp + d] = (f16)(Cm[n*D + d] * w[d])  (zero pad d>=D)
__global__ void mul_bcast_f16(const float* __restrict__ Cm, const float* __restrict__ w,
                              _Float16* __restrict__ out, int T, int D, int Dp)
{
    size_t i = (size_t)blockIdx.x * blockDim.x + threadIdx.x;
    if (i >= (size_t)T * Dp) return;
    int n = (int)(i / Dp), d = (int)(i % Dp);
    out[i] = (d < D) ? (_Float16)(Cm[(size_t)n * D + d] * w[d]) : (_Float16)0.f;
}

// out[r] = dot(A[r,:K1], w) (+ dot(B[r,:K2], w2) if B)
__global__ __launch_bounds__(128)
void rowdot2(const float* __restrict__ A, int lda, const float* __restrict__ w,
             const float* __restrict__ B, int ldb, const float* __restrict__ w2,
             float* __restrict__ out, int K1, int K2)
{
    __shared__ float red[128];
    int r = blockIdx.x;
    float s = 0.f;
    const float* a = A + (size_t)r * lda;
    for (int k = threadIdx.x; k < K1; k += 128) s += a[k] * w[k];
    if (B) {
        const float* bb = B + (size_t)r * ldb;
        for (int k = threadIdx.x; k < K2; k += 128) s += bb[k] * w2[k];
    }
    red[threadIdx.x] = s; __syncthreads();
    for (int st = 64; st > 0; st >>= 1) {
        if (threadIdx.x < st) red[threadIdx.x] += red[threadIdx.x + st];
        __syncthreads();
    }
    if (threadIdx.x == 0) out[r] = red[0];
}

// per-row softmax of sim[r,c]+cwc[r]+qwq[c] over c (=128) -> f16 probs + row max
__global__ __launch_bounds__(128)
void sim_softmax_row(const float* __restrict__ sim, const float* __restrict__ cwc,
                     const float* __restrict__ qwq, _Float16* __restrict__ P,
                     float* __restrict__ rowmax)
{
    __shared__ float red[128];
    int r = blockIdx.x, c = threadIdx.x;
    float v = sim[(size_t)r * 128 + c] + cwc[r] + qwq[c];
    red[c] = v; __syncthreads();
    for (int st = 64; st > 0; st >>= 1) {
        if (c < st) red[c] = fmaxf(red[c], red[c + st]);
        __syncthreads();
    }
    float m = red[0]; __syncthreads();
    float e = __expf(v - m);
    red[c] = e; __syncthreads();
    for (int st = 64; st > 0; st >>= 1) {
        if (c < st) red[c] += red[c + st];
        __syncthreads();
    }
    P[(size_t)r * 128 + c] = (_Float16)(e / red[0]);
    if (c == 0) rowmax[r] = m;
}

// softmax over a length-n vector (single block)
__global__ __launch_bounds__(1024)
void softmax_vec(const float* __restrict__ in, float* __restrict__ out, int n)
{
    __shared__ float red[1024];
    float m = -3.4e38f;
    for (int i = threadIdx.x; i < n; i += 1024) m = fmaxf(m, in[i]);
    red[threadIdx.x] = m; __syncthreads();
    for (int st = 512; st > 0; st >>= 1) {
        if (threadIdx.x < st) red[threadIdx.x] = fmaxf(red[threadIdx.x], red[threadIdx.x + st]);
        __syncthreads();
    }
    m = red[0]; __syncthreads();
    float s = 0.f;
    for (int i = threadIdx.x; i < n; i += 1024) s += __expf(in[i] - m);
    red[threadIdx.x] = s; __syncthreads();
    for (int st = 512; st > 0; st >>= 1) {
        if (threadIdx.x < st) red[threadIdx.x] += red[threadIdx.x + st];
        __syncthreads();
    }
    s = red[0];
    for (int i = threadIdx.x; i < n; i += 1024) out[i] = __expf(in[i] - m) / s;
}

// out[d] = sum_r a[r] * Cm[r,d]
__global__ void weighted_rowsum(const float* __restrict__ a, const float* __restrict__ Cm,
                                float* __restrict__ out, int T, int D)
{
    int d = blockIdx.x * blockDim.x + threadIdx.x;
    if (d >= D) return;
    float s = 0.f;
    for (int r = 0; r < T; ++r) s += a[r] * Cm[(size_t)r * D + d];
    out[d] = s;
}

// qac = [C, c2q, C*c2q, C*q2c]  (f32 + f16, K=2848 already 32-aligned)
__global__ void build_qac(const float* __restrict__ C, const float* __restrict__ c2q,
                          const float* __restrict__ q2c, float* __restrict__ qac,
                          _Float16* __restrict__ qac16, int T)
{
    int i = blockIdx.x * blockDim.x + threadIdx.x;
    if (i >= T * 712) return;
    int n = i / 712, d = i % 712;
    float cv = C[i], c2 = c2q[i], q2 = q2c[d];
    size_t b = (size_t)n * 2848;
    float v0 = cv, v1 = c2, v2 = cv * c2, v3 = cv * q2;
    qac[b + d] = v0;            qac16[b + d] = (_Float16)v0;
    qac[b + 712 + d] = v1;      qac16[b + 712 + d] = (_Float16)v1;
    qac[b + 1424 + d] = v2;     qac16[b + 1424 + d] = (_Float16)v2;
    qac[b + 2136 + d] = v3;     qac16[b + 2136 + d] = (_Float16)v3;
}

// ---------------------------------------------------------------------------
// Host orchestration
// ---------------------------------------------------------------------------
static inline void gemm(hipStream_t s, const _Float16* A, int lda, const _Float16* B, int ldb,
                        const float* bias, float* C, int M, int N, int K, int bt)
{
    dim3 g((N + TN - 1) / TN, (M + TM - 1) / TM);
    if (bt) gemm_wmma<1><<<g, 256, 0, s>>>(A, lda, B, ldb, bias, C, M, N, K);
    else    gemm_wmma<0><<<g, 256, 0, s>>>(A, lda, B, ldb, bias, C, M, N, K);
}
static inline void cvt(hipStream_t s, const float* in, _Float16* out, size_t n)
{
    f32_to_f16<<<(n + 255) / 256, 256, 0, s>>>(in, out, n);
}
static inline void cvt_pad(hipStream_t s, const float* in, _Float16* out, int rows, int K, int Kp)
{
    size_t n = (size_t)rows * Kp;
    f32_to_f16_pad<<<(n + 255) / 256, 256, 0, s>>>(in, out, rows, K, Kp);
}

extern "C" void kernel_launch(void* const* d_in, const int* in_sizes, int n_in,
                              void* d_out, int out_size, void* d_ws, size_t ws_size,
                              hipStream_t stream)
{
    (void)in_sizes; (void)n_in; (void)out_size; (void)ws_size;
    const int TC = 4096, TQ = 128, H = 356, D2 = 712, G = 1424, QC = 2848;
    const int Hp = 384, D2p = 736;   // K-padded dims (multiples of 32)

    const int*   chars_c = (const int*)  d_in[0];
    const int*   chars_q = (const int*)  d_in[1];
    const float* elmo_c  = (const float*)d_in[2];
    const float* elmo_q  = (const float*)d_in[3];
    const float* cemb    = (const float*)d_in[4];
    const float* convw   = (const float*)d_in[5];
    const float* convb   = (const float*)d_in[6];
    const float* hwpw    = (const float*)d_in[7];
    const float* hwpb    = (const float*)d_in[8];
    const float* hwgw    = (const float*)d_in[9];
    const float* hwgb    = (const float*)d_in[10];
    const float* ctxWih  = (const float*)d_in[11];
    const float* ctxWhh  = (const float*)d_in[12];
    const float* ctxb    = (const float*)d_in[13];
    const float* simw    = (const float*)d_in[14];
    const float* m1Wih   = (const float*)d_in[15];
    const float* m1Whh   = (const float*)d_in[16];
    const float* m1b     = (const float*)d_in[17];
    const float* m2Wih   = (const float*)d_in[18];
    const float* m2Whh   = (const float*)d_in[19];
    const float* m2b     = (const float*)d_in[20];
    const float* pWih    = (const float*)d_in[21];
    const float* pWhh    = (const float*)d_in[22];
    const float* pb      = (const float*)d_in[23];
    const float* p1w     = (const float*)d_in[24];
    const float* p2w     = (const float*)d_in[25];
    const float* h0c     = (const float*)d_in[26];
    const float* h0q     = (const float*)d_in[27];
    const float* h0m     = (const float*)d_in[28];
    const float* h0p     = (const float*)d_in[29];
    float* out = (float*)d_out;

    // bump allocator over d_ws
    char* ws = (char*)d_ws;
    size_t off = 0;
    auto alloc = [&](size_t bytes) -> char* {
        off = (off + 255) & ~(size_t)255;
        char* p = ws + off;
        off += bytes;
        return p;
    };
    auto af16 = [&](size_t n) -> _Float16* { return (_Float16*)alloc(n * 2); };
    auto af32 = [&](size_t n) -> float*    { return (float*)   alloc(n * 4); };

    // f16 weight copies (K-padded where used as GEMM operands)
    _Float16* hwp16  = af16((size_t)2 * H * Hp);
    _Float16* hwg16  = af16((size_t)2 * H * Hp);
    _Float16* cWih16 = af16((size_t)2 * G * Hp);
    _Float16* cWhh16 = af16((size_t)2 * G * H);      // scan weights: unpadded
    _Float16* a1Wih16= af16((size_t)2 * G * QC);
    _Float16* a1Whh16= af16((size_t)2 * G * H);
    _Float16* a2Wih16= af16((size_t)2 * G * D2p);
    _Float16* a2Whh16= af16((size_t)2 * G * H);
    _Float16* apWih16= af16((size_t)2 * G * D2p);
    _Float16* apWhh16= af16((size_t)2 * G * H);
    // activations
    float*    cnn_c = af32((size_t)TC * 100);
    float*    cnn_q = af32((size_t)TQ * 100);
    float*    x_c   = af32((size_t)TC * H);
    _Float16* x_c16 = af16((size_t)TC * Hp);
    float*    x_q   = af32((size_t)TQ * H);
    _Float16* x_q16 = af16((size_t)TQ * Hp);
    float*    gz    = af32((size_t)TC * H);
    float*    pz    = af32((size_t)TC * H);
    float*    xWf   = af32((size_t)TC * G);
    float*    xWb   = af32((size_t)TC * G);
    float*    Cmat  = af32((size_t)TC * D2);
    float*    Qmat  = af32((size_t)TQ * D2);
    _Float16* Q16   = af16((size_t)TQ * D2p);
    _Float16* CW16  = af16((size_t)TC * D2p);
    float*    cwc   = af32(TC);
    float*    qwq   = af32(TQ);
    float*    sim   = af32((size_t)TC * TQ);
    _Float16* P16   = af16((size_t)TC * TQ);
    float*    rowm  = af32(TC);
    float*    aat   = af32(TC);
    float*    c2q   = af32((size_t)TC * D2);
    float*    q2c   = af32(D2);
    float*    qac   = af32((size_t)TC * QC);
    _Float16* qac16 = af16((size_t)TC * QC);
    float*    M1    = af32((size_t)TC * D2);
    _Float16* M116  = af16((size_t)TC * D2p);
    float*    Mm    = af32((size_t)TC * D2);
    _Float16* Mm16  = af16((size_t)TC * D2p);
    float*    Mp    = af32((size_t)TC * D2);
    float*    lg1   = af32(TC);
    float*    lg2   = af32(TC);

    // ---- weight conversions ----
    cvt_pad(stream, hwpw,   hwp16,   2 * H, H, Hp);
    cvt_pad(stream, hwgw,   hwg16,   2 * H, H, Hp);
    cvt_pad(stream, ctxWih, cWih16,  2 * G, H, Hp);
    cvt    (stream, ctxWhh, cWhh16,  (size_t)2 * G * H);
    cvt_pad(stream, m1Wih,  a1Wih16, 2 * G, QC, QC);
    cvt    (stream, m1Whh,  a1Whh16, (size_t)2 * G * H);
    cvt_pad(stream, m2Wih,  a2Wih16, 2 * G, D2, D2p);
    cvt    (stream, m2Whh,  a2Whh16, (size_t)2 * G * H);
    cvt_pad(stream, pWih,   apWih16, 2 * G, D2, D2p);
    cvt    (stream, pWhh,   apWhh16, (size_t)2 * G * H);

    // ---- char CNN + concat with elmo ----
    char_cnn<<<(TC * 100 + 127) / 128, 128, 0, stream>>>(chars_c, cemb, convw, convb, cnn_c, TC);
    char_cnn<<<(TQ * 100 + 127) / 128, 128, 0, stream>>>(chars_q, cemb, convw, convb, cnn_q, TQ);
    build_x<<<(TC * Hp + 255) / 256, 256, 0, stream>>>(cnn_c, elmo_c, x_c, x_c16, TC, Hp);
    build_x<<<(TQ * Hp + 255) / 256, 256, 0, stream>>>(cnn_q, elmo_q, x_q, x_q16, TQ, Hp);

    // ---- highway (2 layers) for ctx and qry ----
    for (int i = 0; i < 2; ++i) {
        gemm(stream, x_c16, Hp, hwg16 + (size_t)i * H * Hp, Hp, hwgb + i * H, gz, TC, H, Hp, 1);
        gemm(stream, x_c16, Hp, hwp16 + (size_t)i * H * Hp, Hp, hwpb + i * H, pz, TC, H, Hp, 1);
        hw_combine<<<(TC * H + 255) / 256, 256, 0, stream>>>(gz, pz, x_c, x_c16, TC * H, Hp);
    }
    for (int i = 0; i < 2; ++i) {
        gemm(stream, x_q16, Hp, hwg16 + (size_t)i * H * Hp, Hp, hwgb + i * H, gz, TQ, H, Hp, 1);
        gemm(stream, x_q16, Hp, hwp16 + (size_t)i * H * Hp, Hp, hwpb + i * H, pz, TQ, H, Hp, 1);
        hw_combine<<<(TQ * H + 255) / 256, 256, 0, stream>>>(gz, pz, x_q, x_q16, TQ * H, Hp);
    }

    // ---- ctx BiLSTM -> C, Q ----
    gemm(stream, x_c16, Hp, cWih16,                  Hp, ctxb,     xWf, TC, G, Hp, 1);
    gemm(stream, x_c16, Hp, cWih16 + (size_t)G * Hp, Hp, ctxb + G, xWb, TC, G, Hp, 1);
    lstm_scan<<<2, 384, 0, stream>>>(xWf, xWb, cWhh16, h0c, Cmat, TC);
    gemm(stream, x_q16, Hp, cWih16,                  Hp, ctxb,     xWf, TQ, G, Hp, 1);
    gemm(stream, x_q16, Hp, cWih16 + (size_t)G * Hp, Hp, ctxb + G, xWb, TQ, G, Hp, 1);
    lstm_scan<<<2, 384, 0, stream>>>(xWf, xWb, cWhh16, h0q, Qmat, TQ);
    cvt_pad(stream, Qmat, Q16, TQ, D2, D2p);

    // ---- attention ----
    rowdot2<<<TC, 128, 0, stream>>>(Cmat, D2, simw,      nullptr, 0, nullptr, cwc, D2, 0);
    rowdot2<<<TQ, 128, 0, stream>>>(Qmat, D2, simw + D2, nullptr, 0, nullptr, qwq, D2, 0);
    mul_bcast_f16<<<((size_t)TC * D2p + 255) / 256, 256, 0, stream>>>(Cmat, simw + 2 * D2, CW16, TC, D2, D2p);
    gemm(stream, CW16, D2p, Q16, D2p, nullptr, sim, TC, TQ, D2p, 1);   // (C*w_cq) @ Q^T
    sim_softmax_row<<<TC, 128, 0, stream>>>(sim, cwc, qwq, P16, rowm);
    gemm(stream, P16, TQ, Q16, D2p, nullptr, c2q, TC, D2, TQ, 0);      // P @ Q
    softmax_vec<<<1, 1024, 0, stream>>>(rowm, aat, TC);
    weighted_rowsum<<<(D2 + 255) / 256, 256, 0, stream>>>(aat, Cmat, q2c, TC, D2);
    build_qac<<<(TC * D2 + 255) / 256, 256, 0, stream>>>(Cmat, c2q, q2c, qac, qac16, TC);

    // ---- mod1 ----
    gemm(stream, qac16, QC, a1Wih16,                  QC, m1b,     xWf, TC, G, QC, 1);
    gemm(stream, qac16, QC, a1Wih16 + (size_t)G * QC, QC, m1b + G, xWb, TC, G, QC, 1);
    lstm_scan<<<2, 384, 0, stream>>>(xWf, xWb, a1Whh16, h0m, M1, TC);
    cvt_pad(stream, M1, M116, TC, D2, D2p);
    // ---- mod2 ----
    gemm(stream, M116, D2p, a2Wih16,                   D2p, m2b,     xWf, TC, G, D2p, 1);
    gemm(stream, M116, D2p, a2Wih16 + (size_t)G * D2p, D2p, m2b + G, xWb, TC, G, D2p, 1);
    lstm_scan<<<2, 384, 0, stream>>>(xWf, xWb, a2Whh16, h0m + 2 * H, Mm, TC);
    cvt_pad(stream, Mm, Mm16, TC, D2, D2p);

    // ---- pos1 ----
    rowdot2<<<TC, 128, 0, stream>>>(qac, QC, p1w, Mm, D2, p1w + QC, lg1, QC, D2);
    softmax_vec<<<1, 1024, 0, stream>>>(lg1, out, TC);

    // ---- pos BiLSTM -> Mp, pos2 ----
    gemm(stream, Mm16, D2p, apWih16,                   D2p, pb,     xWf, TC, G, D2p, 1);
    gemm(stream, Mm16, D2p, apWih16 + (size_t)G * D2p, D2p, pb + G, xWb, TC, G, D2p, 1);
    lstm_scan<<<2, 384, 0, stream>>>(xWf, xWb, apWhh16, h0p, Mp, TC);
    rowdot2<<<TC, 128, 0, stream>>>(qac, QC, p2w, Mp, D2, p2w + QC, lg2, QC, D2);
    softmax_vec<<<1, 1024, 0, stream>>>(lg2, out + TC, TC);
}
